// GPTModel_52072183497489
// MI455X (gfx1250) — matrix operand order repaired and checked by reading, not compiled, not run on hardware
//
#include <hip/hip_runtime.h>

// ---------------- model constants ----------------
constexpr int Bb  = 2;
constexpr int T   = 1024;
constexpr int Cd  = 1024;
constexpr int H   = 16;
constexpr int Ln  = 6;
constexpr int Vv  = 32000;
constexpr int Pp  = 4;
constexpr int HD  = Cd / H;         // 64
constexpr int M   = Bb * T;         // 2048 rows of activations
constexpr int FF  = Pp * Cd;        // 4096

// ---------------- WMMA types ----------------
typedef __attribute__((ext_vector_type(16))) __bf16 v16bf;
typedef __attribute__((ext_vector_type(8)))  float  v8f;

__device__ inline __bf16 f2bf(float f) {
  union { float f; unsigned u; } v; v.f = f;
  unsigned r = v.u + 0x7fffu + ((v.u >> 16) & 1u);   // RNE
  unsigned short h = (unsigned short)(r >> 16);
  return __builtin_bit_cast(__bf16, h);
}

__device__ inline v8f wmma_bf16(v16bf a, v16bf b, v8f c) {
  return __builtin_amdgcn_wmma_f32_16x16x32_bf16(
      /*neg_a=*/false, a, /*neg_b=*/false, b,
      /*c_mod=*/(short)0, c, /*reuse_a=*/false, /*reuse_b=*/false);
}

// A-fragment (16x32 bf16, MxK) from row-major LDS tile with row stride `ld`.
__device__ inline v16bf load_a_frag(const __bf16* base, int r0, int ld, int lane) {
  int row = r0 + (lane & 15);
  int kb  = (lane < 16) ? 0 : 16;
  const __bf16* p = base + row * ld + kb;
  v16bf a;
#pragma unroll
  for (int i = 0; i < 8; ++i) {
    int k = 2 * i;
    a[2 * i]     = p[k];
    a[2 * i + 1] = p[k + 1];
  }
  return a;
}

// B-fragment (32x16 bf16, KxN) from LDS stored [n][k] with row stride `ld`.
__device__ inline v16bf load_b_frag(const __bf16* base, int n0, int ld, int lane) {
  int col = n0 + (lane & 15);
  int kb  = (lane < 16) ? 0 : 16;
  const __bf16* p = base + col * ld + kb;
  v16bf b;
#pragma unroll
  for (int i = 0; i < 8; ++i) { b[2 * i] = p[2 * i]; b[2 * i + 1] = p[2 * i + 1]; }
  return b;
}

__device__ inline float gelu_f(float x) {
  return 0.5f * x * (1.0f + erff(x * 0.70710678118654752f));
}

// Async global->LDS copy of 16 bytes (CDNA5 path) with fallback.
#if defined(__has_builtin)
#if __has_builtin(__builtin_amdgcn_global_load_async_to_lds_b128)
#define HAVE_ASYNC_LDS 1
#endif
#endif

typedef int v4i __attribute__((vector_size(16)));
typedef __attribute__((address_space(1))) v4i* g_v4i_p;
typedef __attribute__((address_space(3))) v4i* l_v4i_p;

__device__ inline void copy16_g2l(const __bf16* g, __bf16* l) {
#ifdef HAVE_ASYNC_LDS
  __builtin_amdgcn_global_load_async_to_lds_b128((g_v4i_p)g, (l_v4i_p)l, 0, 0);
#else
  *(uint4*)l = *(const uint4*)g;
#endif
}

__device__ inline void wait_async_lds() {
#ifdef HAVE_ASYNC_LDS
#if __has_builtin(__builtin_amdgcn_s_wait_asynccnt)
  __builtin_amdgcn_s_wait_asynccnt(0);
#else
  asm volatile("s_wait_asynccnt 0x0" ::: "memory");
#endif
#endif
}

// ---------------- kernels ----------------

__global__ __launch_bounds__(256) void embed_kernel(
    const int* __restrict__ tokens, const float* __restrict__ table,
    float* __restrict__ x) {
  int row = blockIdx.x;                 // 0..M-1
  int tok = tokens[row];
  const float* src = table + (size_t)tok * Cd;
  float* dst = x + (size_t)row * Cd;
  for (int c = threadIdx.x; c < Cd; c += 256) dst[c] = src[c];
}

__global__ __launch_bounds__(256) void rmsnorm_kernel(
    const float* __restrict__ x, const float* __restrict__ g,
    __bf16* __restrict__ out) {
  int row = blockIdx.x;
  const float* xr = x + (size_t)row * Cd;
  float s = 0.f;
  for (int c = threadIdx.x; c < Cd; c += 256) { float v = xr[c]; s += v * v; }
  __shared__ float red[256];
  red[threadIdx.x] = s; __syncthreads();
  for (int off = 128; off > 0; off >>= 1) {
    if (threadIdx.x < off) red[threadIdx.x] += red[threadIdx.x + off];
    __syncthreads();
  }
  float inv = rsqrtf(red[0] / (float)Cd + 1.1920928955078125e-7f);
  __bf16* o = out + (size_t)row * Cd;
  for (int c = threadIdx.x; c < Cd; c += 256) o[c] = f2bf(xr[c] * inv * g[c]);
}

__global__ __launch_bounds__(256) void f32_to_bf16_kernel(
    const float* __restrict__ in, __bf16* __restrict__ out, int n) {
  int i = blockIdx.x * 256 + threadIdx.x;
  if (i < n) out[i] = f2bf(in[i]);
}

// RoPE + head split: qkv [M][3C] fp32 -> qb,kb [B,H,T,HD] bf16 (rotated),
// vbT [B,H,HD,T] bf16 (transposed for PV B-fragments).
__global__ __launch_bounds__(256) void rope_split_kernel(
    const float* __restrict__ qkv, __bf16* __restrict__ qb,
    __bf16* __restrict__ kb, __bf16* __restrict__ vbT) {
  int row = blockIdx.x;                  // b*T + t
  int b = row / T, t = row % T;
  const float* qr = qkv + (size_t)row * (3 * Cd);
  for (int p = threadIdx.x; p < Cd / 2; p += 256) {
    float theta = __powf(10000.0f, -2.0f * (float)p / (float)Cd);
    float sn, cs; __sincosf((float)t * theta, &sn, &cs);
    int c0 = 2 * p, c1 = 2 * p + 1;
    int h = c0 / HD;
    int d0 = c0 % HD, d1 = d0 + 1;
    size_t base = ((size_t)(b * H + h) * T + t) * HD;
    float q0 = qr[c0], q1 = qr[c1];
    qb[base + d0] = f2bf(q0 * cs - q1 * sn);
    qb[base + d1] = f2bf(q1 * cs + q0 * sn);
    float k0 = qr[Cd + c0], k1 = qr[Cd + c1];
    kb[base + d0] = f2bf(k0 * cs - k1 * sn);
    kb[base + d1] = f2bf(k1 * cs + k0 * sn);
    size_t vb = (size_t)(b * H + h) * HD * T;
    vbT[vb + (size_t)d0 * T + t] = f2bf(qr[2 * Cd + c0]);
    vbT[vb + (size_t)d1 * T + t] = f2bf(qr[2 * Cd + c1]);
  }
}

// Generic WMMA GEMM: out = act(A(bf16)[MxK] @ Bw(f32)[KxN] + bias) (+resid)
// Block: 256 threads = 8 waves (2x4), 64x128 tile, BK=32; wave = 32x32 subtile.
constexpr int LDT = 40;  // padded LDS row stride (bf16 elems)

__global__ __launch_bounds__(256) void gemm_wmma_kernel(
    const __bf16* __restrict__ A, const float* __restrict__ Bw,
    const float* __restrict__ bias, const float* __restrict__ resid,
    float* __restrict__ outF, __bf16* __restrict__ outB,
    int Mm, int Nn, int Kk, int act) {
  __shared__ __bf16 As[64 * LDT];
  __shared__ __bf16 Bs[128 * LDT];
  int tid = threadIdx.x, lane = tid & 31, w = tid >> 5;
  int wm = w >> 2, wn = w & 3;               // 2 x 4 wave grid
  int bm = blockIdx.y * 64, bn = blockIdx.x * 128;
  v8f acc00 = {}, acc01 = {}, acc10 = {}, acc11 = {};
  for (int k0 = 0; k0 < Kk; k0 += 32) {
    // A tile 64x32 bf16: pure copy, 16B per thread, async to LDS
    {
      int r = tid >> 2, seg = tid & 3;       // 4 x 16B segs per 64B row
      copy16_g2l(A + (size_t)(bm + r) * Kk + k0 + seg * 8,
                 As + r * LDT + seg * 8);
    }
    // B tile 32x128 f32 -> bf16, stored [n][k] so K-pairs are contiguous
    for (int i = tid; i < 32 * 128; i += 256) {
      int kk = i >> 7, n = i & 127;
      Bs[n * LDT + kk] = f2bf(Bw[(size_t)(k0 + kk) * Nn + bn + n]);
    }
    // prefetch next K-tile of the streamed fp32 weights
    if (k0 + 32 < Kk) {
      int kk = tid >> 3, n16 = (tid & 7) * 16;
      __builtin_prefetch(&Bw[(size_t)(k0 + 32 + kk) * Nn + bn + n16], 0, 1);
    }
    wait_async_lds();
    __syncthreads();
    v16bf a0 = load_a_frag(As, wm * 32,      LDT, lane);
    v16bf a1 = load_a_frag(As, wm * 32 + 16, LDT, lane);
    v16bf b0 = load_b_frag(Bs, wn * 32,      LDT, lane);
    v16bf b1 = load_b_frag(Bs, wn * 32 + 16, LDT, lane);
    acc00 = wmma_bf16(a0, b0, acc00);
    acc10 = wmma_bf16(a1, b0, acc10);
    acc01 = wmma_bf16(a0, b1, acc01);
    acc11 = wmma_bf16(a1, b1, acc11);
    __syncthreads();
  }
#pragma unroll
  for (int ni = 0; ni < 2; ++ni) {
    int ng = bn + wn * 32 + ni * 16 + (lane & 15);
    float bv = bias ? bias[ng] : 0.0f;
#pragma unroll
    for (int mi = 0; mi < 2; ++mi) {
      v8f acc = (mi == 0) ? (ni == 0 ? acc00 : acc01)
                          : (ni == 0 ? acc10 : acc11);
      int mbase = bm + wm * 32 + mi * 16 + ((lane < 16) ? 0 : 8);
#pragma unroll
      for (int r = 0; r < 8; ++r) {
        int m = mbase + r;
        float v = acc[r] + bv;
        if (act == 1) v = gelu_f(v);
        if (resid) v += resid[(size_t)m * Nn + ng];
        if (outF) outF[(size_t)m * Nn + ng] = v;
        if (outB) outB[(size_t)m * Nn + ng] = f2bf(v);
      }
    }
  }
}

// Flash-style causal attention: one wave per 16-query tile of one (b,h).
__global__ __launch_bounds__(256) void attn_wmma_kernel(
    const __bf16* __restrict__ qb, const __bf16* __restrict__ kb,
    const __bf16* __restrict__ vbT, __bf16* __restrict__ aout) {
  __shared__ __bf16 Pl[8 * 16 * LDT];          // per-wave P staging
  int lane = threadIdx.x & 31, w = threadIdx.x >> 5;
  int gw = blockIdx.x * 8 + w;
  int qt = gw & (T / 16 - 1);
  int bh = gw >> 6;
  const __bf16* Q  = qb  + (size_t)bh * T * HD;
  const __bf16* Kp = kb  + (size_t)bh * T * HD;
  const __bf16* Vt = vbT + (size_t)bh * HD * T;
  __bf16* Pw = Pl + w * 16 * LDT;

  v16bf qf0, qf1;
  {
    int row = qt * 16 + (lane & 15);
    int kbse = (lane < 16) ? 0 : 16;
#pragma unroll
    for (int h2 = 0; h2 < 2; ++h2) {
      const __bf16* p = Q + (size_t)row * HD + h2 * 32 + kbse;
      v16bf a;
#pragma unroll
      for (int i = 0; i < 8; ++i) {
        int k = 2 * i;
        a[2 * i] = p[k]; a[2 * i + 1] = p[k + 1];
      }
      if (h2 == 0) qf0 = a; else qf1 = a;
    }
  }

  v8f o0 = {}, o1 = {}, o2 = {}, o3 = {};
  float mrow[8], lrow[8];
#pragma unroll
  for (int r = 0; r < 8; ++r) { mrow[r] = -INFINITY; lrow[r] = 0.f; }
  const float scale = 0.125f;                  // 1/sqrt(64)
  int mloc = (lane < 16) ? 0 : 8;

  for (int j = 0; j <= qt; j += 2) {
    bool has2 = (j + 1) <= qt;
    v8f s0 = {}, s1 = {};
    {
      int key0 = j * 16 + (lane & 15);
      int kbse = (lane < 16) ? 0 : 16;
#pragma unroll
      for (int h2 = 0; h2 < 2; ++h2) {
        const __bf16* p0 = Kp + (size_t)key0 * HD + h2 * 32 + kbse;
        v16bf b0;
#pragma unroll
        for (int i = 0; i < 8; ++i) { b0[2 * i] = p0[2 * i]; b0[2 * i + 1] = p0[2 * i + 1]; }
        s0 = wmma_bf16(h2 ? qf1 : qf0, b0, s0);
        if (has2) {
          const __bf16* p1 = Kp + (size_t)(key0 + 16) * HD + h2 * 32 + kbse;
          v16bf b1;
#pragma unroll
          for (int i = 0; i < 8; ++i) { b1[2 * i] = p1[2 * i]; b1[2 * i + 1] = p1[2 * i + 1]; }
          s1 = wmma_bf16(h2 ? qf1 : qf0, b1, s1);
        }
      }
    }
    int n0 = j * 16 + (lane & 15);
    int n1 = n0 + 16;
#pragma unroll
    for (int r = 0; r < 8; ++r) {
      int m = qt * 16 + r + mloc;
      float a0 = s0[r] * scale; if (n0 > m) a0 = -INFINITY;
      float a1 = has2 ? (s1[r] * scale) : -INFINITY; if (n1 > m) a1 = -INFINITY;
      float mx = fmaxf(a0, a1);
#pragma unroll
      for (int off = 1; off < 16; off <<= 1) mx = fmaxf(mx, __shfl_xor(mx, off));
      float mnew = fmaxf(mrow[r], mx);
      float corr = __expf(mrow[r] - mnew);
      float p0 = __expf(a0 - mnew);
      float p1 = __expf(a1 - mnew);
      float rs = p0 + p1;
#pragma unroll
      for (int off = 1; off < 16; off <<= 1) rs += __shfl_xor(rs, off);
      lrow[r] = lrow[r] * corr + rs;
      mrow[r] = mnew;
      o0[r] *= corr; o1[r] *= corr; o2[r] *= corr; o3[r] *= corr;
      int ml = r + mloc;
      Pw[ml * LDT + (lane & 15)]      = f2bf(p0);
      Pw[ml * LDT + 16 + (lane & 15)] = f2bf(p1);
    }
    v16bf pa = load_a_frag(Pw, 0, LDT, lane);
    int kclamp = has2 ? ((lane < 16) ? 0 : 16) : 0;
#pragma unroll
    for (int nt = 0; nt < 4; ++nt) {
      int hd = nt * 16 + (lane & 15);
      const __bf16* p = Vt + (size_t)hd * T + j * 16 + kclamp;
      v16bf vf;
#pragma unroll
      for (int i = 0; i < 8; ++i) { vf[2 * i] = p[2 * i]; vf[2 * i + 1] = p[2 * i + 1]; }
      v8f acc = (nt == 0) ? o0 : (nt == 1) ? o1 : (nt == 2) ? o2 : o3;
      acc = wmma_bf16(pa, vf, acc);
      if (nt == 0) o0 = acc; else if (nt == 1) o1 = acc; else if (nt == 2) o2 = acc; else o3 = acc;
    }
  }

  int b = bh >> 4, h = bh & 15;
#pragma unroll
  for (int r = 0; r < 8; ++r) {
    float inv = 1.0f / lrow[r];
    int tglob = qt * 16 + r + mloc;
    size_t base = ((size_t)(b * T + tglob)) * Cd + h * HD + (lane & 15);
    aout[base +  0] = f2bf(o0[r] * inv);
    aout[base + 16] = f2bf(o1[r] * inv);
    aout[base + 32] = f2bf(o2[r] * inv);
    aout[base + 48] = f2bf(o3[r] * inv);
  }
}

// ---------------- host orchestration ----------------
extern "C" void kernel_launch(void* const* d_in, const int* in_sizes, int n_in,
                              void* d_out, int out_size, void* d_ws, size_t ws_size,
                              hipStream_t stream) {
  const int*   tokens = (const int*)d_in[0];
  const float* embed  = (const float*)d_in[1];
  const float* w_attn = (const float*)d_in[2];
  const float* b_attn = (const float*)d_in[3];
  const float* w_proj = (const float*)d_in[4];
  const float* b_proj = (const float*)d_in[5];
  const float* g1     = (const float*)d_in[6];
  const float* g2     = (const float*)d_in[7];
  const float* w_ff1  = (const float*)d_in[8];
  const float* b_ff1  = (const float*)d_in[9];
  const float* w_ff2  = (const float*)d_in[10];
  const float* b_ff2  = (const float*)d_in[11];
  const float* w_out  = (const float*)d_in[12];
  float* logits = (float*)d_out;

  char* ws = (char*)d_ws;
  size_t off = 0;
  auto alloc = [&](size_t bytes) { void* p = ws + off; off += (bytes + 255) & ~(size_t)255; return p; };
  float*  x    = (float*) alloc((size_t)M * Cd * 4);
  __bf16* hbf  = (__bf16*)alloc((size_t)M * Cd * 2);
  float*  qkv  = (float*) alloc((size_t)M * 3 * Cd * 4);
  __bf16* qb   = (__bf16*)alloc((size_t)M * Cd * 2);
  __bf16* kbuf = (__bf16*)alloc((size_t)M * Cd * 2);
  __bf16* vbT  = (__bf16*)alloc((size_t)M * Cd * 2);
  __bf16* abf  = (__bf16*)alloc((size_t)M * Cd * 2);
  __bf16* ffb  = (__bf16*)alloc((size_t)M * FF * 2);

  embed_kernel<<<M, 256, 0, stream>>>(tokens, embed, x);

  for (int l = 0; l < Ln; ++l) {
    const float* wa = w_attn + (size_t)l * Cd * 3 * Cd;
    const float* ba = b_attn + (size_t)l * 3 * Cd;
    const float* wp = w_proj + (size_t)l * Cd * Cd;
    const float* bp = b_proj + (size_t)l * Cd;
    const float* w1 = w_ff1  + (size_t)l * Cd * FF;
    const float* bb1= b_ff1  + (size_t)l * FF;
    const float* w2 = w_ff2  + (size_t)l * FF * Cd;
    const float* bb2= b_ff2  + (size_t)l * Cd;

    rmsnorm_kernel<<<M, 256, 0, stream>>>(x, g1 + (size_t)l * Cd, hbf);
    gemm_wmma_kernel<<<dim3(3 * Cd / 128, M / 64), 256, 0, stream>>>(
        hbf, wa, ba, nullptr, qkv, nullptr, M, 3 * Cd, Cd, 0);
    rope_split_kernel<<<M, 256, 0, stream>>>(qkv, qb, kbuf, vbT);
    attn_wmma_kernel<<<(Bb * H * (T / 16)) / 8, 256, 0, stream>>>(qb, kbuf, vbT, abf);
    gemm_wmma_kernel<<<dim3(Cd / 128, M / 64), 256, 0, stream>>>(
        abf, wp, bp, x, x, nullptr, M, Cd, Cd, 0);
    rmsnorm_kernel<<<M, 256, 0, stream>>>(x, g2 + (size_t)l * Cd, hbf);
    gemm_wmma_kernel<<<dim3(FF / 128, M / 64), 256, 0, stream>>>(
        hbf, w1, bb1, nullptr, nullptr, ffb, M, FF, Cd, 1);
    gemm_wmma_kernel<<<dim3(Cd / 128, M / 64), 256, 0, stream>>>(
        ffb, w2, bb2, x, x, nullptr, M, Cd, FF, 0);
  }

  f32_to_bf16_kernel<<<(M * Cd + 255) / 256, 256, 0, stream>>>(x, hbf, M * Cd);
  gemm_wmma_kernel<<<dim3(Vv / 128, M / 64), 256, 0, stream>>>(
      hbf, w_out, nullptr, nullptr, logits, nullptr, M, Vv, Cd, 0);
}
